// LinearAttention_45346264711335
// MI455X (gfx1250) — compile-verified, matrix operand order checked by reading, not run on hardware
//
#include <hip/hip_runtime.h>
#include <hip/hip_bf16.h>
#include <math.h>

typedef __bf16 bf16;
typedef __attribute__((ext_vector_type(16))) __bf16 v16bf;
typedef __attribute__((ext_vector_type(8)))  float  v8f;

#define EPSV 1e-6f

// ---------------------------------------------------------------------------
// WMMA helpers (gfx1250: v_wmma_f32_16x16x32_bf16, wave32)
// A fragment (16x32 bf16): element e -> K = (e/8)*16 + (e%8) + (lane/16)*8,
// pairs contiguous -> compiler emits 2x ds_load_b128 per fragment.
// For P = X * Y^T with X[M][K], Y[N][K] row-major, A and B loads are identical.
// C/D (16x16 f32): n = lane%16, vgpr r -> m = r + (lane/16)*8.
// ---------------------------------------------------------------------------

__device__ __forceinline__ v8f wmma_bf16(v16bf a, v16bf b, v8f c) {
  return __builtin_amdgcn_wmma_f32_16x16x32_bf16(false, a, false, b, (short)0, c,
                                                 false, false);
}

__device__ __forceinline__ v16bf load_frag_bf(const bf16* base, int row0, int ld, int k0) {
  const int lane = threadIdx.x & 31;
  const bf16* p = base + (size_t)(row0 + (lane & 15)) * ld + k0 + ((lane >> 4) << 3);
  v16bf r;
#pragma unroll
  for (int pp = 0; pp < 8; ++pp) {
    const int k = ((pp >> 2) << 4) + ((pp & 3) << 1);   // 0,2,4,6,16,18,20,22
    const unsigned u = *(const unsigned*)(p + k);
    r[2 * pp]     = ((const bf16*)&u)[0];
    r[2 * pp + 1] = ((const bf16*)&u)[1];
  }
  return r;
}

__device__ __forceinline__ v16bf load_frag_f32(const float* base, int row0, int ld, int k0) {
  const int lane = threadIdx.x & 31;
  const float* p = base + (size_t)(row0 + (lane & 15)) * ld + k0 + ((lane >> 4) << 3);
  v16bf r;
#pragma unroll
  for (int e = 0; e < 16; ++e) {
    const int k = ((e >> 3) << 4) + (e & 7);
    r[e] = (bf16)p[k];
  }
  return r;
}

__device__ __forceinline__ void pack4(bf16* dst, float4 f) {
  union { uint2 u; bf16 h[4]; } p;
  p.h[0] = (bf16)f.x; p.h[1] = (bf16)f.y; p.h[2] = (bf16)f.z; p.h[3] = (bf16)f.w;
  *(uint2*)dst = p.u;                                   // ds_store_b64
}

// elu(v)+1, exact and branchless: v>0 -> v+exp(0)=v+1 ; v<=0 -> 0+exp(v)
__device__ __forceinline__ float phi(float v) {
  return fmaxf(v, 0.f) + __expf(fminf(v, 0.f));
}

// ---------------------------------------------------------------------------
// Kernel 1: qkv = phi_or_id(x @ w_qkv^T), bf16 out.  M=2048, N=3072, K=1024.
// Block tile 128x64; per K-step each wave: 1 A frag reused over 4 WMMA.
// ---------------------------------------------------------------------------
__global__ __launch_bounds__(256) void qkv_phi_gemm(const float* __restrict__ x,
                                                    const float* __restrict__ w,
                                                    bf16* __restrict__ out) {
  __shared__ bf16 As[128 * 32];
  __shared__ bf16 Bs[64 * 32];
  const int tid = threadIdx.x, wv = tid >> 5, lane = tid & 31;
  const int bx = blockIdx.x, by = blockIdx.y;
  const bool doPhi = (bx < 32);                  // q,k columns; v columns identity

  v8f acc[4];
#pragma unroll
  for (int tn = 0; tn < 4; ++tn)
#pragma unroll
    for (int q = 0; q < 8; ++q) acc[tn][q] = 0.f;

  for (int kk = 0; kk < 1024; kk += 32) {
#pragma unroll
    for (int r = 0; r < 4; ++r) {                // A: 128x32 via float4
      const int e = tid + 256 * r;               // 0..1023
      const int i = e >> 3, c = (e & 7) * 4;
      pack4(As + i * 32 + c,
            *(const float4*)(x + (size_t)(by * 128 + i) * 1024 + kk + c));
    }
#pragma unroll
    for (int r = 0; r < 2; ++r) {                // B: 64x32 via float4
      const int e = tid + 256 * r;               // 0..511
      const int j = e >> 3, c = (e & 7) * 4;
      pack4(Bs + j * 32 + c,
            *(const float4*)(w + (size_t)(bx * 64 + j) * 1024 + kk + c));
    }
    if (kk + 32 < 1024) {                        // pull next 128B panel lines
      if (tid < 128)
        __builtin_prefetch(x + (size_t)(by * 128 + tid) * 1024 + kk + 32, 0, 3);
      else if (tid < 192)
        __builtin_prefetch(w + (size_t)(bx * 64 + (tid - 128)) * 1024 + kk + 32, 0, 3);
    }
    __syncthreads();
    v16bf a = load_frag_bf(As, wv * 16, 32, 0);
#pragma unroll
    for (int tn = 0; tn < 4; ++tn) {
      v16bf b = load_frag_bf(Bs, tn * 16, 32, 0);
      acc[tn] = wmma_bf16(a, b, acc[tn]);
    }
    __syncthreads();
  }

  const int n = lane & 15, mh = (lane >> 4) * 8;
#pragma unroll
  for (int tn = 0; tn < 4; ++tn)
#pragma unroll
    for (int rr = 0; rr < 8; ++rr) {
      const int t = by * 128 + wv * 16 + rr + mh;
      const int j = bx * 64 + tn * 16 + n;
      float v = acc[tn][rr];
      if (doPhi) v = phi(v);
      out[(size_t)t * 3072 + j] = (bf16)v;
    }
}

// ---------------------------------------------------------------------------
// Kernel 2: chunked causal linear attention, one workgroup per head.
// Global loads are software-pipelined through registers: chunk c+1's loads
// issue right after chunk c's staging barrier and overlap all WMMA phases,
// hiding ~1us HBM latency per chunk that would otherwise serialize (32x).
// ---------------------------------------------------------------------------
__global__ __launch_bounds__(256) void linattn_chunk(const bf16* __restrict__ qkv,
                                                     bf16* __restrict__ xo) {
  __shared__ bf16 Qs[64 * 64];    // [i][d]
  __shared__ bf16 Ks[64 * 64];    // [j][d]
  __shared__ bf16 KsT[64 * 64];   // [d][j]
  __shared__ bf16 VsT[64 * 64];   // [f][j]
  __shared__ bf16 Ss[64 * 64];    // masked scores [i][j]
  __shared__ float ST[64 * 64];   // state^T [f][d]
  __shared__ float zS[64];
  __shared__ float denomS[64];

  const int tid = threadIdx.x, wv = tid >> 5, lane = tid & 31;
  const int h = blockIdx.x;

  for (int e = tid; e < 64 * 64; e += 256) ST[e] = 0.f;
  if (tid < 64) zS[tid] = 0.f;

  uint2 pq[4], pk[4], pv[4];                     // in-flight chunk (regs)
  auto load_chunk = [&](int c) {
#pragma unroll
    for (int r = 0; r < 4; ++r) {
      const int e = tid + 256 * r;               // 0..1023
      const int i = e >> 4, g = (e & 15) * 4;
      const size_t rb = (size_t)(c * 64 + i) * 3072 + h * 64 + g;
      pq[r] = *(const uint2*)(qkv + rb);
      pk[r] = *(const uint2*)(qkv + rb + 1024);
      pv[r] = *(const uint2*)(qkv + rb + 2048);
    }
  };
  auto store_chunk = [&]() {
#pragma unroll
    for (int r = 0; r < 4; ++r) {
      const int e = tid + 256 * r;
      const int i = e >> 4, g = (e & 15) * 4;
      *(uint2*)(Qs + i * 64 + g) = pq[r];
      *(uint2*)(Ks + i * 64 + g) = pk[r];
      const bf16* kp = (const bf16*)&pk[r];
      const bf16* vp = (const bf16*)&pv[r];
#pragma unroll
      for (int t = 0; t < 4; ++t) {
        KsT[(g + t) * 64 + i] = kp[t];
        VsT[(g + t) * 64 + i] = vp[t];
      }
    }
  };

  load_chunk(0);
  v8f o[2];

  for (int c = 0; c < 32; ++c) {
    // ---- P0: stage in-flight chunk into LDS ------------------------------
    store_chunk();
    __syncthreads();
    if (c < 31) load_chunk(c + 1);               // prefetch next (overlaps P1-P3)

    // ---- P1: scores + inter part of O (shared Q fragment) ----------------
#pragma unroll
    for (int rep = 0; rep < 2; ++rep) {
      const int tt = wv * 2 + rep, tm = tt >> 2, tn = tt & 3;
      v8f sc;
#pragma unroll
      for (int q = 0; q < 8; ++q) { sc[q] = 0.f; o[rep][q] = 0.f; }
#pragma unroll
      for (int k0 = 0; k0 < 64; k0 += 32) {
        v16bf aq = load_frag_bf(Qs, tm * 16, 64, k0);
        v16bf bk = load_frag_bf(Ks, tn * 16, 64, k0);
        sc = wmma_bf16(aq, bk, sc);
        v16bf bs = load_frag_f32(ST, tn * 16, 64, k0);   // S^T[f][d]
        o[rep] = wmma_bf16(aq, bs, o[rep]);
      }
      const int n = lane & 15, mh = (lane >> 4) * 8;
#pragma unroll
      for (int rr = 0; rr < 8; ++rr) {
        const int i = tm * 16 + rr + mh, j = tn * 16 + n;
        Ss[i * 64 + j] = (bf16)((j <= i) ? sc[rr] : 0.f);  // causal mask
      }
    }
    __syncthreads();

    // ---- P2: O += Ss @ V ; denominators (use z_prev) ---------------------
#pragma unroll
    for (int rep = 0; rep < 2; ++rep) {
      const int tt = wv * 2 + rep, tm = tt >> 2, tn = tt & 3;
#pragma unroll
      for (int k0 = 0; k0 < 64; k0 += 32) {
        v16bf a = load_frag_bf(Ss, tm * 16, 64, k0);
        v16bf b = load_frag_bf(VsT, tn * 16, 64, k0);
        o[rep] = wmma_bf16(a, b, o[rep]);
      }
    }
    if (tid < 64) {
      float acc = EPSV;
      for (int d = 0; d < 64; ++d) acc += (float)Qs[tid * 64 + d] * zS[d];
      for (int j = 0; j < 64; ++j) acc += (float)Ss[tid * 64 + j];
      denomS[tid] = acc;
    }
    __syncthreads();

    // ---- P3: S^T += V^T K ; z update ; normalized output -----------------
#pragma unroll
    for (int rep = 0; rep < 2; ++rep) {
      const int tt = wv * 2 + rep, tm = tt >> 2, tn = tt & 3;
      const int n = lane & 15, mh = (lane >> 4) * 8;
      v8f st;
#pragma unroll
      for (int rr = 0; rr < 8; ++rr)
        st[rr] = ST[(tm * 16 + rr + mh) * 64 + tn * 16 + n];
#pragma unroll
      for (int k0 = 0; k0 < 64; k0 += 32) {                // P[f][d] = V^T K
        v16bf a = load_frag_bf(VsT, tm * 16, 64, k0);
        v16bf b = load_frag_bf(KsT, tn * 16, 64, k0);
        st = wmma_bf16(a, b, st);
      }
#pragma unroll
      for (int rr = 0; rr < 8; ++rr)
        ST[(tm * 16 + rr + mh) * 64 + tn * 16 + n] = st[rr];
#pragma unroll
      for (int rr = 0; rr < 8; ++rr) {
        const int i = tm * 16 + rr + mh, f = tn * 16 + n;
        const float v = o[rep][rr] / denomS[i];
        xo[(size_t)(c * 64 + i) * 1024 + h * 64 + f] = (bf16)v;
      }
    }
    if (tid < 64) {
      float s = 0.f;
      for (int j = 0; j < 64; ++j) s += (float)KsT[tid * 64 + j];
      zS[tid] += s;
    }
    __syncthreads();                             // LDS consumed; next stage OK
  }
}

// ---------------------------------------------------------------------------
// Kernel 3: out = xo @ w_out^T.  M=2048, N=1024, K=1024.  128x64 tile, fp32 out.
// ---------------------------------------------------------------------------
__global__ __launch_bounds__(256) void out_gemm(const bf16* __restrict__ xo,
                                                const float* __restrict__ w,
                                                float* __restrict__ out) {
  __shared__ bf16 As[128 * 32];
  __shared__ bf16 Bs[64 * 32];
  const int tid = threadIdx.x, wv = tid >> 5, lane = tid & 31;
  const int bx = blockIdx.x, by = blockIdx.y;

  v8f acc[4];
#pragma unroll
  for (int tn = 0; tn < 4; ++tn)
#pragma unroll
    for (int q = 0; q < 8; ++q) acc[tn][q] = 0.f;

  for (int kk = 0; kk < 1024; kk += 32) {
#pragma unroll
    for (int r = 0; r < 4; ++r) {                // A: 128x32 bf16 copy (b64)
      const int e = tid + 256 * r;
      const int i = e >> 3, c = (e & 7) * 4;
      *(uint2*)(As + i * 32 + c) =
          *(const uint2*)(xo + (size_t)(by * 128 + i) * 1024 + kk + c);
    }
#pragma unroll
    for (int r = 0; r < 2; ++r) {                // B: 64x32 fp32 -> bf16
      const int e = tid + 256 * r;
      const int j = e >> 3, c = (e & 7) * 4;
      pack4(Bs + j * 32 + c,
            *(const float4*)(w + (size_t)(bx * 64 + j) * 1024 + kk + c));
    }
    if (kk + 32 < 1024) {
      if (tid < 128)
        __builtin_prefetch(xo + (size_t)(by * 128 + tid) * 1024 + kk + 32, 0, 3);
      else if (tid < 192)
        __builtin_prefetch(w + (size_t)(bx * 64 + (tid - 128)) * 1024 + kk + 32, 0, 3);
    }
    __syncthreads();
    v16bf a = load_frag_bf(As, wv * 16, 32, 0);
#pragma unroll
    for (int tn = 0; tn < 4; ++tn) {
      v16bf b = load_frag_bf(Bs, tn * 16, 32, 0);
      acc[tn] = wmma_bf16(a, b, acc[tn]);
    }
    __syncthreads();
  }

  const int n = lane & 15, mh = (lane >> 4) * 8;
#pragma unroll
  for (int tn = 0; tn < 4; ++tn)
#pragma unroll
    for (int rr = 0; rr < 8; ++rr) {
      const int t = by * 128 + wv * 16 + rr + mh;
      const int j = bx * 64 + tn * 16 + n;
      out[(size_t)t * 1024 + j] = acc[tn][rr];
    }
}

// ---------------------------------------------------------------------------
extern "C" void kernel_launch(void* const* d_in, const int* in_sizes, int n_in,
                              void* d_out, int out_size, void* d_ws, size_t ws_size,
                              hipStream_t stream) {
  const float* x     = (const float*)d_in[0];   // [2048,1024]
  const float* w_qkv = (const float*)d_in[1];   // [3072,1024]
  const float* w_out = (const float*)d_in[2];   // [1024,1024]
  float* out = (float*)d_out;                   // [2048,1024]

  bf16* qkvbf = (bf16*)d_ws;                                        // [2048,3072]
  bf16* xobf  = (bf16*)((char*)d_ws + (size_t)2048 * 3072 * 2);     // [2048,1024]

  dim3 blk(256);
  qkv_phi_gemm<<<dim3(48, 16), blk, 0, stream>>>(x, w_qkv, qkvbf);
  linattn_chunk<<<dim3(16), blk, 0, stream>>>(qkvbf, xobf);
  out_gemm<<<dim3(16, 16), blk, 0, stream>>>(xobf, w_out, out);
}
